// BigramLanguageModel_38663295599421
// MI455X (gfx1250) — compile-verified, matrix operand order checked by reading, not run on hardware
//
#include <hip/hip_runtime.h>
#include <hip/hip_bf16.h>
#include <math.h>

// ---------------------------------------------------------------------------
// GPT forward for MI455X (gfx1250, wave32). All GEMMs run through
// v_wmma_f32_16x16x32_f16 (f16 inputs converted on the fly, f32 accumulate).
// LDS tiles for BOTH A and B are kept in the ISA fragment layout so every
// per-lane fragment fetch is two contiguous ds_load_b128s.
// ---------------------------------------------------------------------------

typedef __attribute__((ext_vector_type(16))) _Float16 v16h;
typedef __attribute__((ext_vector_type(8)))  _Float16 v8h;
typedef __attribute__((ext_vector_type(8)))  float    v8f;

#define BM 128
#define BN 64
#define BKK 32
#define LDK 40   // padded row stride in halves: 80 bytes (16B-aligned, bank-skewed)

// ---------------------------------------------------------------------------
// Generic GEMM: C[M,N] = A[M,K] * B[K,N] (+bias) (+relu)
//   A row-major, leading dim lda.
//   bTrans==0: B element (k,n) at B[k*ldb + n]  (row-major [K,N])
//   bTrans==1: B element (k,n) at B[n*ldb + k]  (row-major [N,K] -> A*B^T)
// Block = 256 threads = 8 wave32s; 128x64 C-tile per block; each wave owns
// 32 rows x 32 cols = 4 WMMA accumulators, K stepped by 32.
// Requires M%128==0, N%64==0, K%32==0 (true for every GEMM in this model).
// ---------------------------------------------------------------------------
__global__ __launch_bounds__(256)
void gemm_f16_wmma(const float* __restrict__ A, int lda,
                   const float* __restrict__ B, int ldb, int bTrans,
                   const float* __restrict__ bias,
                   float* __restrict__ C, int ldc,
                   int M, int N, int K, int doRelu)
{
    __shared__ _Float16 As [BM][LDK];   // A tile, row-major [m][k]
    __shared__ _Float16 BsT[BN][LDK];   // B tile TRANSPOSED: [n][k]

    const int tid   = threadIdx.x;
    const int m0    = blockIdx.y * BM;
    const int n0    = blockIdx.x * BN;
    const int w     = tid >> 5;          // wave id 0..7
    const int lane  = tid & 31;
    const int row16 = lane & 15;         // M index (A frag) / N index (B frag)
    const int grp   = lane >> 4;         // K-half select per ISA 16-bit layout
    const int waveM = (w >> 1) * 32;     // 4 wave rows of 32
    const int waveN = (w & 1) * 32;      // 2 wave cols of 32

    // A staging: one thread stages 16 contiguous halves of one row
    const int arow = tid >> 1;           // 0..127
    const int acol = (tid & 1) * 16;     // 0 or 16
    // B staging: one thread stages one (n, k-octet)
    const int bn  = tid & 63;            // 0..63
    const int bk8 = (tid >> 6) * 8;      // 0,8,16,24

    v8f c00 = {}, c01 = {}, c10 = {}, c11 = {};

    for (int k0 = 0; k0 < K; k0 += BKK) {
        // ---- stage A tile (f32 -> f16), contiguous 16-half chunk ----
        {
            v8h lo, hi;
            #pragma unroll
            for (int i = 0; i < 8; ++i) {
                lo[i] = (_Float16)A[(size_t)(m0 + arow) * lda + (k0 + acol + i)];
                hi[i] = (_Float16)A[(size_t)(m0 + arow) * lda + (k0 + acol + 8 + i)];
            }
            *(v8h*)&As[arow][acol]     = lo;
            *(v8h*)&As[arow][acol + 8] = hi;
        }

        // ---- stage B tile transposed: BsT[n][k] ----
        {
            v8h t;
            if (!bTrans) {
                #pragma unroll
                for (int i = 0; i < 8; ++i)   // coalesced across lanes (n fast)
                    t[i] = (_Float16)B[(size_t)(k0 + bk8 + i) * ldb + (n0 + bn)];
            } else {
                #pragma unroll
                for (int i = 0; i < 8; ++i)   // contiguous per thread
                    t[i] = (_Float16)B[(size_t)(n0 + bn) * ldb + (k0 + bk8 + i)];
            }
            *(v8h*)&BsT[bn][bk8] = t;
        }

        // prefetch next A K-tile into cache (global_prefetch_b8)
        if (k0 + BKK < K)
            __builtin_prefetch(&A[(size_t)(m0 + arow) * lda + (k0 + BKK + acol)], 0, 1);

        __syncthreads();

        // ---- fragments: each lane reads two contiguous 8-half chunks ----
        // lane<16 holds K = {0..7, 16..23}; lane>=16 holds K = {8..15, 24..31}
        v16h a0, a1, b0, b1;
        #pragma unroll
        for (int e = 0; e < 16; ++e) {
            const int p  = e >> 1;
            const int s  = e & 1;
            const int kk = ((p < 4) ? (p * 2) : (16 + (p - 4) * 2)) + grp * 8 + s;
            a0[e] = As [waveM +      row16][kk];
            a1[e] = As [waveM + 16 + row16][kk];
            b0[e] = BsT[waveN +      row16][kk];
            b1[e] = BsT[waveN + 16 + row16][kk];
        }

        c00 = __builtin_amdgcn_wmma_f32_16x16x32_f16(false, a0, false, b0,
                                                     (short)0, c00, false, false);
        c01 = __builtin_amdgcn_wmma_f32_16x16x32_f16(false, a0, false, b1,
                                                     (short)0, c01, false, false);
        c10 = __builtin_amdgcn_wmma_f32_16x16x32_f16(false, a1, false, b0,
                                                     (short)0, c10, false, false);
        c11 = __builtin_amdgcn_wmma_f32_16x16x32_f16(false, a1, false, b1,
                                                     (short)0, c11, false, false);
        __syncthreads();
    }

    // ---- epilogue: C layout = 8 VGPRs, lane half selects M+8 ----
    #pragma unroll
    for (int su = 0; su < 2; ++su) {
        #pragma unroll
        for (int sv = 0; sv < 2; ++sv) {
            const v8f& c = su ? (sv ? c11 : c10) : (sv ? c01 : c00);
            const int n  = n0 + waveN + sv * 16 + row16;
            float badd = bias ? bias[n] : 0.f;
            #pragma unroll
            for (int r = 0; r < 8; ++r) {
                const int m = m0 + waveM + su * 16 + r + grp * 8;
                float vv = c[r] + badd;
                if (doRelu) vv = fmaxf(vv, 0.f);
                C[(size_t)m * ldc + n] = vv;
            }
        }
    }
}

// ---------------------------------------------------------------------------
// x[b,t,:] = tok_emb[idx[b,t],:] + pos_emb[t,:]
// ---------------------------------------------------------------------------
__global__ __launch_bounds__(256)
void embed_kernel(const int* __restrict__ idx, const float* __restrict__ tok,
                  const float* __restrict__ pos, float* __restrict__ x,
                  int T, int D, size_t n)
{
    const size_t stride = (size_t)gridDim.x * blockDim.x;
    for (size_t i = (size_t)blockIdx.x * blockDim.x + threadIdx.x; i < n; i += stride) {
        const size_t row = i / D;
        const int d = (int)(i % D);
        const int t = (int)(row % T);
        x[i] = tok[(size_t)idx[row] * D + d] + pos[(size_t)t * D + d];
    }
}

// ---------------------------------------------------------------------------
// Row LayerNorm: y = (x-mean)*rsqrt(var+eps)*g + b, one block per row.
// ---------------------------------------------------------------------------
__global__ __launch_bounds__(256)
void layernorm_kernel(const float* __restrict__ x, const float* __restrict__ g,
                      const float* __restrict__ b, float* __restrict__ y, int D)
{
    const int row = blockIdx.x;
    const float* xr = x + (size_t)row * D;
    float* yr = y + (size_t)row * D;
    __shared__ float red[256];

    float s = 0.f;
    for (int i = threadIdx.x; i < D; i += 256) s += xr[i];
    red[threadIdx.x] = s; __syncthreads();
    for (int o = 128; o > 0; o >>= 1) {
        if (threadIdx.x < o) red[threadIdx.x] += red[threadIdx.x + o];
        __syncthreads();
    }
    const float mean = red[0] / D;
    __syncthreads();

    s = 0.f;
    for (int i = threadIdx.x; i < D; i += 256) { float d = xr[i] - mean; s += d * d; }
    red[threadIdx.x] = s; __syncthreads();
    for (int o = 128; o > 0; o >>= 1) {
        if (threadIdx.x < o) red[threadIdx.x] += red[threadIdx.x + o];
        __syncthreads();
    }
    const float rstd = rsqrtf(red[0] / D + 1e-5f);

    for (int i = threadIdx.x; i < D; i += 256)
        yr[i] = (xr[i] - mean) * rstd * g[i] + b[i];
}

// ---------------------------------------------------------------------------
// Reference quirk: softmax over the QUERY axis of scores[t,s] (axis=2 of
// [B,H,Tq,Tk]).  One block per column s; valid region is t in [s, T)
// (causal tril mask makes t<s be -inf -> 0 after softmax).
// ---------------------------------------------------------------------------
__global__ __launch_bounds__(256)
void col_softmax_kernel(float* __restrict__ scores, int T, float scale)
{
    const int s = blockIdx.x;
    __shared__ float red[256];

    float mx = -INFINITY;
    for (int t = s + threadIdx.x; t < T; t += 256)
        mx = fmaxf(mx, scores[(size_t)t * T + s] * scale);
    red[threadIdx.x] = mx; __syncthreads();
    for (int o = 128; o > 0; o >>= 1) {
        if (threadIdx.x < o) red[threadIdx.x] = fmaxf(red[threadIdx.x], red[threadIdx.x + o]);
        __syncthreads();
    }
    mx = red[0]; __syncthreads();

    float sum = 0.f;
    for (int t = s + threadIdx.x; t < T; t += 256)
        sum += __expf(scores[(size_t)t * T + s] * scale - mx);
    red[threadIdx.x] = sum; __syncthreads();
    for (int o = 128; o > 0; o >>= 1) {
        if (threadIdx.x < o) red[threadIdx.x] += red[threadIdx.x + o];
        __syncthreads();
    }
    const float inv = 1.0f / red[0];

    for (int t = threadIdx.x; t < T; t += 256) {
        const size_t a = (size_t)t * T + s;
        scores[a] = (t >= s) ? __expf(scores[a] * scale - mx) * inv : 0.0f;
    }
}

__global__ __launch_bounds__(256)
void add_kernel(const float* __restrict__ a, const float* __restrict__ b,
                float* __restrict__ out, size_t n)
{
    const size_t stride = (size_t)gridDim.x * blockDim.x;
    for (size_t i = (size_t)blockIdx.x * blockDim.x + threadIdx.x; i < n; i += stride)
        out[i] = a[i] + b[i];
}

__global__ void zero_scalar_kernel(float* p) { *p = 0.f; }

// loss += -(logit[tgt] - max - log(sum exp)) / N, one block per row.
__global__ __launch_bounds__(256)
void loss_kernel(const float* __restrict__ logits, const int* __restrict__ targets,
                 float* __restrict__ loss, int V, float invN)
{
    const int row = blockIdx.x;
    const float* lr = logits + (size_t)row * V;
    __shared__ float red[256];

    float mx = -INFINITY;
    for (int i = threadIdx.x; i < V; i += 256) mx = fmaxf(mx, lr[i]);
    red[threadIdx.x] = mx; __syncthreads();
    for (int o = 128; o > 0; o >>= 1) {
        if (threadIdx.x < o) red[threadIdx.x] = fmaxf(red[threadIdx.x], red[threadIdx.x + o]);
        __syncthreads();
    }
    mx = red[0]; __syncthreads();

    float sum = 0.f;
    for (int i = threadIdx.x; i < V; i += 256) sum += __expf(lr[i] - mx);
    red[threadIdx.x] = sum; __syncthreads();
    for (int o = 128; o > 0; o >>= 1) {
        if (threadIdx.x < o) red[threadIdx.x] += red[threadIdx.x + o];
        __syncthreads();
    }

    if (threadIdx.x == 0) {
        const int tgt = targets[row];
        const float logp = lr[tgt] - mx - __logf(red[0]);
        atomicAdd(loss, -logp * invN);
    }
}

// ---------------------------------------------------------------------------
// Host orchestration
// ---------------------------------------------------------------------------
static inline void launch_gemm(const float* A, int lda, const float* B, int ldb,
                               int bTrans, const float* bias, float* C, int ldc,
                               int M, int N, int K, int relu, hipStream_t stream)
{
    dim3 grid((unsigned)(N / BN), (unsigned)(M / BM));
    gemm_f16_wmma<<<grid, 256, 0, stream>>>(A, lda, B, ldb, bTrans, bias, C, ldc,
                                            M, N, K, relu);
}

extern "C" void kernel_launch(void* const* d_in, const int* in_sizes, int n_in,
                              void* d_out, int out_size, void* d_ws, size_t ws_size,
                              hipStream_t stream)
{
    (void)in_sizes; (void)n_in; (void)out_size; (void)ws_size;

    const int Bb = 2, T = 2048, V = 32000, D = 1024, H = 16, L = 8, HS = 64, F = 4096;
    const int BT = Bb * T;                 // 4096 rows
    const float scale = 1.0f / 32.0f;      // D^-0.5 (quirk: scales by D, not hs)

    const int*   idx     = (const int*)  d_in[0];
    const int*   targets = (const int*)  d_in[1];
    const float* tok_emb = (const float*)d_in[2];
    const float* pos_emb = (const float*)d_in[3];
    const float* wq      = (const float*)d_in[4];
    const float* wk      = (const float*)d_in[5];
    const float* wv      = (const float*)d_in[6];
    const float* wproj   = (const float*)d_in[7];
    const float* bproj   = (const float*)d_in[8];
    const float* ln1_g   = (const float*)d_in[9];
    const float* ln1_b   = (const float*)d_in[10];
    const float* ln2_g   = (const float*)d_in[11];
    const float* ln2_b   = (const float*)d_in[12];
    const float* w1      = (const float*)d_in[13];
    const float* b1      = (const float*)d_in[14];
    const float* w2      = (const float*)d_in[15];
    const float* b2      = (const float*)d_in[16];
    const float* lnf_g   = (const float*)d_in[17];
    const float* lnf_b   = (const float*)d_in[18];
    const float* wlm     = (const float*)d_in[19];
    const float* blm     = (const float*)d_in[20];

    float* out = (float*)d_out;            // [BT*V] logits, then [1] loss

    // workspace layout (floats)
    const size_t SZ_ACT = (size_t)BT * D;          // 4096*1024
    const size_t SZ_SC  = (size_t)T * T;           // 2048*2048
    float* ws = (float*)d_ws;
    float* x      = ws;                    ws += SZ_ACT;
    float* h      = ws;                    ws += SZ_ACT;   // LN output (h / h2)
    float* q      = ws;                    ws += SZ_ACT;   // also proj/mlp2 out
    float* k      = ws;                    ws += SZ_ACT;
    float* v      = ws;                    ws += SZ_ACT;
    float* o      = ws;                    ws += SZ_ACT;
    float* scores = ws;                    ws += SZ_SC;
    float* f1     = ws;                    /* ws += (size_t)BT * F; */

    // ---- embedding ----
    embed_kernel<<<4096, 256, 0, stream>>>(idx, tok_emb, pos_emb, x, T, D, SZ_ACT);

    for (int l = 0; l < L; ++l) {
        // ---- LN1 ----
        layernorm_kernel<<<BT, 256, 0, stream>>>(x, ln1_g + (size_t)l * D,
                                                 ln1_b + (size_t)l * D, h, D);

        // ---- QKV per head: [BT,D] x [D,HS] ----
        for (int hd = 0; hd < H; ++hd) {
            const size_t woff = ((size_t)l * H + hd) * (size_t)D * HS;
            launch_gemm(h, D, wq + woff, HS, 0, nullptr, q + hd * HS, D, BT, HS, D, 0, stream);
            launch_gemm(h, D, wk + woff, HS, 0, nullptr, k + hd * HS, D, BT, HS, D, 0, stream);
            launch_gemm(h, D, wv + woff, HS, 0, nullptr, v + hd * HS, D, BT, HS, D, 0, stream);
        }

        // ---- attention per (b, head), serialized on stream (scores reused) ----
        for (int b = 0; b < Bb; ++b) {
            for (int hd = 0; hd < H; ++hd) {
                const float* qb = q + (size_t)b * T * D + hd * HS;
                const float* kb = k + (size_t)b * T * D + hd * HS;
                const float* vb = v + (size_t)b * T * D + hd * HS;
                float*       ob = o + (size_t)b * T * D + hd * HS;
                // scores = Q * K^T  [T,T]
                launch_gemm(qb, D, kb, D, /*bTrans=*/1, nullptr, scores, T,
                            T, T, HS, 0, stream);
                // softmax over the QUERY axis (reference quirk) + causal mask
                col_softmax_kernel<<<T, 256, 0, stream>>>(scores, T, scale);
                // o = W * V  [T,HS]
                launch_gemm(scores, T, vb, D, 0, nullptr, ob, D, T, HS, T, 0, stream);
            }
        }

        // ---- output projection: o @ wproj + bproj -> q (scratch) ----
        launch_gemm(o, D, wproj + (size_t)l * D * D, D, 0,
                    bproj + (size_t)l * D, q, D, BT, D, D, 0, stream);
        // residual from post-LN activations (as in original): x = h + proj
        add_kernel<<<4096, 256, 0, stream>>>(h, q, x, SZ_ACT);

        // ---- LN2 ----
        layernorm_kernel<<<BT, 256, 0, stream>>>(x, ln2_g + (size_t)l * D,
                                                 ln2_b + (size_t)l * D, h, D);

        // ---- MLP: relu(h @ w1 + b1) @ w2 + b2 ----
        launch_gemm(h, D, w1 + (size_t)l * D * F, F, 0,
                    b1 + (size_t)l * F, f1, F, BT, F, D, /*relu=*/1, stream);
        launch_gemm(f1, F, w2 + (size_t)l * F * D, D, 0,
                    b2 + (size_t)l * D, q, D, BT, D, F, 0, stream);
        // x = h2 + f
        add_kernel<<<4096, 256, 0, stream>>>(h, q, x, SZ_ACT);
    }

    // ---- final LN ----
    layernorm_kernel<<<BT, 256, 0, stream>>>(x, lnf_g, lnf_b, h, D);

    // ---- LM head: logits straight into d_out ----
    launch_gemm(h, D, wlm, V, 0, blm, out, V, BT, V, D, 0, stream);

    // ---- loss ----
    float* loss = out + (size_t)BT * V;
    zero_scalar_kernel<<<1, 1, 0, stream>>>(loss);
    loss_kernel<<<BT, 256, 0, stream>>>(out, targets, loss, V, 1.0f / (float)BT);
}
